// SmallWorldEncoder_6330781794651
// MI455X (gfx1250) — matrix operand order repaired and checked by reading, not compiled, not run on hardware
//
#include <hip/hip_runtime.h>
#include <hip/hip_bf16.h>

typedef __attribute__((ext_vector_type(16))) _Float16 v16h;
typedef __attribute__((ext_vector_type(8)))  float    v8f;
typedef __attribute__((ext_vector_type(4)))  _Float16 v4h;

#define OUT_CH   256
#define EMB_DIM  128
#define NUM_CONN 3276          // int(256*128*0.1)
#define LN_EPS   1e-5f
#define M_BLK    128           // rows per block
#define LSTRIDE  136           // 128 halves + 8 pad (kills LDS bank conflicts)

union FragH { v16h h; uint4 q[2]; };
union Acc   { v8f  v; float f[8]; };

// ---------------------------------------------------------------------------
// Prep: W_comb(f16) = W_reg + 0.1 * scatter_set(values)  ;  W_out -> f16
// Serial scatter in one thread reproduces jax .at[].set last-write-wins.
// ---------------------------------------------------------------------------
__global__ __launch_bounds__(256) void swe_prep(
    const float* __restrict__ W_reg,
    const long long* __restrict__ row_idx,
    const long long* __restrict__ col_idx,
    const float* __restrict__ values,
    const float* __restrict__ W_out,
    float* __restrict__ Wc32,
    _Float16* __restrict__ Wc16,
    _Float16* __restrict__ Wout16)
{
    const int tid = threadIdx.x;
    for (int i = tid; i < EMB_DIM * OUT_CH; i += 256) Wc32[i] = W_reg[i];
    __syncthreads();
    if (tid == 0) {
        for (int i = 0; i < NUM_CONN; ++i) {
            int r = (int)row_idx[i];
            int c = (int)col_idx[i];
            // base + 0.1*v each time -> duplicates: last write wins (matches .set)
            Wc32[r * OUT_CH + c] = W_reg[r * OUT_CH + c] + 0.1f * values[i];
        }
    }
    __syncthreads();
    for (int i = tid; i < EMB_DIM * OUT_CH; i += 256)  Wc16[i]  = (_Float16)Wc32[i];
    for (int i = tid; i < EMB_DIM * EMB_DIM; i += 256) Wout16[i] = (_Float16)W_out[i];
}

// ---------------------------------------------------------------------------
// Main fused kernel: gather -> GEMM1(WMMA f16) -> +b_reg -> LN -> GEMM2(WMMA)
//                    -> +b_out -> relu -> store f32
// ---------------------------------------------------------------------------
__global__ __launch_bounds__(256, 1) void swe_main(
    const long long* __restrict__ sol,     // (64,2048,2) int64
    const float* __restrict__ embds,       // (100000,256) f32
    const _Float16* __restrict__ Wc16,     // (128,256) f16, row-major [o][c]
    const _Float16* __restrict__ Wout16,   // (128,128) f16, row-major [e][d]
    const float* __restrict__ b_reg,
    const float* __restrict__ ln_gamma,
    const float* __restrict__ ln_beta,
    const float* __restrict__ b_out,
    float* __restrict__ out)               // (131072,128) f32
{
    __shared__ _Float16 lds_x[M_BLK * LSTRIDE];   // 34816 B, reused for hn
    __shared__ int vtx[M_BLK];

    const int tid  = threadIdx.x;
    const int lane = tid & 31;
    const int w    = tid >> 5;              // wave id 0..7 -> M tile
    const int row_base = blockIdx.x * M_BLK;

    if (tid < M_BLK) vtx[tid] = (int)sol[(long long)(row_base + tid) * 2];
    __syncthreads();

    const int hi   = (lane >= 16) ? 1 : 0;  // upper half of wave
    const int l15  = lane & 15;
    const int mrow = w * 16 + l15;          // A-fragment source row (lane's M)

    Acc acc[8];
    #pragma unroll
    for (int nt = 0; nt < 8; ++nt)
        #pragma unroll
        for (int j = 0; j < 8; ++j) acc[nt].f[j] = 0.0f;

    // ---------------- GEMM1: h = x @ Wc^T, K = 256 in two LDS halves -------
    for (int kh = 0; kh < 2; ++kh) {
        // cooperative gather + f32->f16 convert of 128 rows x 128 cols
        for (int it = tid; it < M_BLK * 32; it += 256) {
            const int r  = it >> 5;
            const int ch = it & 31;
            const float4 s =
                *(const float4*)(embds + (long long)vtx[r] * OUT_CH + kh * 128 + ch * 4);
            v4h d;
            d.x = (_Float16)s.x; d.y = (_Float16)s.y;
            d.z = (_Float16)s.z; d.w = (_Float16)s.w;
            *(v4h*)(&lds_x[r * LSTRIDE + ch * 4]) = d;
        }
        __syncthreads();

        #pragma unroll
        for (int ks = 0; ks < 4; ++ks) {
            // A fragment (contiguous K, matches B): halves 0-15 @ K=base+16*hi
            FragH a;
            const int koff = ks * 32 + hi * 16;
            a.q[0] = *(const uint4*)(&lds_x[mrow * LSTRIDE + koff]);
            a.q[1] = *(const uint4*)(&lds_x[mrow * LSTRIDE + koff + 8]);
            const int kg = kh * 128 + ks * 32 + hi * 16;  // global K for B
            #pragma unroll
            for (int nt = 0; nt < 8; ++nt) {
                const int n = nt * 16 + l15;
                FragH b;
                const uint4* wp = (const uint4*)(Wc16 + n * OUT_CH + kg);
                b.q[0] = wp[0];
                b.q[1] = wp[1];
                acc[nt].v = __builtin_amdgcn_wmma_f32_16x16x32_f16(
                    false, a.h, false, b.h, (short)0, acc[nt].v, false, false);
            }
        }
        __syncthreads();
    }

    // ---------------- +b_reg, LayerNorm over N=128 (in-wave) ---------------
    #pragma unroll
    for (int nt = 0; nt < 8; ++nt) {
        const float br = b_reg[nt * 16 + l15];
        #pragma unroll
        for (int j = 0; j < 8; ++j) acc[nt].f[j] += br;
    }

    float mu[8], rstd[8];
    #pragma unroll
    for (int r = 0; r < 8; ++r) {
        float s = 0.0f;
        #pragma unroll
        for (int nt = 0; nt < 8; ++nt) s += acc[nt].f[r];
        s += __shfl_xor(s, 1, 32);
        s += __shfl_xor(s, 2, 32);
        s += __shfl_xor(s, 4, 32);
        s += __shfl_xor(s, 8, 32);   // reduction stays inside each 16-lane half
        mu[r] = s * (1.0f / EMB_DIM);
        float q = 0.0f;
        #pragma unroll
        for (int nt = 0; nt < 8; ++nt) {
            const float d = acc[nt].f[r] - mu[r];
            q += d * d;
        }
        q += __shfl_xor(q, 1, 32);
        q += __shfl_xor(q, 2, 32);
        q += __shfl_xor(q, 4, 32);
        q += __shfl_xor(q, 8, 32);
        rstd[r] = rsqrtf(q * (1.0f / EMB_DIM) + LN_EPS);
    }

    // normalize, gamma/beta, write hn (f16) into LDS (reuse lds_x), zero acc
    #pragma unroll
    for (int nt = 0; nt < 8; ++nt) {
        const int n = nt * 16 + l15;
        const float g  = ln_gamma[n];
        const float bt = ln_beta[n];
        #pragma unroll
        for (int r = 0; r < 8; ++r) {
            const int m = w * 16 + r + hi * 8;     // C-layout row for (reg r, lane)
            const float hn = (acc[nt].f[r] - mu[r]) * rstd[r] * g + bt;
            lds_x[m * LSTRIDE + n] = (_Float16)hn;
            acc[nt].f[r] = 0.0f;
        }
    }
    __syncthreads();

    // ---------------- GEMM2: out = relu(hn @ Wout^T + b_out), K = 128 ------
    #pragma unroll
    for (int ks = 0; ks < 4; ++ks) {
        FragH a;
        const int koff = ks * 32 + hi * 16;
        a.q[0] = *(const uint4*)(&lds_x[mrow * LSTRIDE + koff]);
        a.q[1] = *(const uint4*)(&lds_x[mrow * LSTRIDE + koff + 8]);
        const int kg = ks * 32 + hi * 16;
        #pragma unroll
        for (int nt = 0; nt < 8; ++nt) {
            const int n = nt * 16 + l15;
            FragH b;
            const uint4* wp = (const uint4*)(Wout16 + n * EMB_DIM + kg);
            b.q[0] = wp[0];
            b.q[1] = wp[1];
            acc[nt].v = __builtin_amdgcn_wmma_f32_16x16x32_f16(
                false, a.h, false, b.h, (short)0, acc[nt].v, false, false);
        }
    }

    #pragma unroll
    for (int nt = 0; nt < 8; ++nt) {
        const int n = nt * 16 + l15;
        const float bo = b_out[n];
        #pragma unroll
        for (int r = 0; r < 8; ++r) {
            const int m = w * 16 + r + hi * 8;
            const float v = acc[nt].f[r] + bo;
            out[(long long)(row_base + m) * EMB_DIM + n] = fmaxf(v, 0.0f);
        }
    }
}

// ---------------------------------------------------------------------------
extern "C" void kernel_launch(void* const* d_in, const int* in_sizes, int n_in,
                              void* d_out, int out_size, void* d_ws, size_t ws_size,
                              hipStream_t stream)
{
    // setup_inputs() order:
    const long long* solutions = (const long long*)d_in[0]; // (64,2048,2) i64
    const float*     embds     = (const float*)d_in[1];     // (100000,256)
    const float*     W_reg     = (const float*)d_in[2];     // (128,256)
    const float*     b_reg     = (const float*)d_in[3];     // (128,)
    const long long* row_idx   = (const long long*)d_in[4]; // (3276,)
    const long long* col_idx   = (const long long*)d_in[5];
    const float*     values    = (const float*)d_in[6];
    const float*     ln_gamma  = (const float*)d_in[7];
    const float*     ln_beta   = (const float*)d_in[8];
    const float*     W_out     = (const float*)d_in[9];     // (128,128)
    const float*     b_out     = (const float*)d_in[10];

    char* ws = (char*)d_ws;
    _Float16* Wc16   = (_Float16*)(ws);                 //  64 KB
    _Float16* Wout16 = (_Float16*)(ws + 65536);         //  32 KB
    float*    Wc32   = (float*)(ws + 98304);            // 128 KB scratch

    swe_prep<<<1, 256, 0, stream>>>(W_reg, row_idx, col_idx, values, W_out,
                                    Wc32, Wc16, Wout16);

    const int total_rows = 64 * 2048;                   // 131072
    swe_main<<<total_rows / M_BLK, 256, 0, stream>>>(
        solutions, embds, Wc16, Wout16,
        b_reg, ln_gamma, ln_beta, b_out, (float*)d_out);
}
